// L2LOptimizer_5076651344477
// MI455X (gfx1250) — matrix operand order repaired and checked by reading, not compiled, not run on hardware
//
#include <hip/hip_runtime.h>
#include <hip/hip_bf16.h>

typedef __attribute__((ext_vector_type(16))) _Float16 v16h;
typedef __attribute__((ext_vector_type(8)))  _Float16 v8h;
typedef __attribute__((ext_vector_type(8)))  float    v8f;

#define HDIM  64
#define NGATE 256   // 4*H

static __device__ __forceinline__ v8f wmma_f16(v16h a, v16h b, v8f c) {
  // D = A(16x32 f16) * B(32x16 f16) + C(16x16 f32)
  return __builtin_amdgcn_wmma_f32_16x16x32_f16(false, a, false, b, (short)0, c, false, false);
}

static __device__ __forceinline__ v8f bcast8(float v) {
  v8f c;
#pragma unroll
  for (int r = 0; r < 8; ++r) c[r] = v;
  return c;
}

static __device__ __forceinline__ v16h cat8(v8h lo, v8h hi) {
  return __builtin_shufflevector(lo, hi, 0,1,2,3,4,5,6,7,8,9,10,11,12,13,14,15);
}

// A-fragment (16x32 f16) from per-wave LDS tile stored row-major [16][64] halves.
// lanes 0-15:  M=lane,    elems 0..7 = K kb..kb+7,   elems 8..15 = K kb+16..kb+23
// lanes 16-31: M=lane-16, elems 0..7 = K kb+8..+15,  elems 8..15 = K kb+24..+31
static __device__ __forceinline__ v16h loadA_lds(const _Float16* s, int kbase, int lh, int hs) {
  const _Float16* rp = s + lh * HDIM;
  const int k0 = kbase + 8 * hs;
  const int k1 = kbase + 16 + 8 * hs;
  v8h lo = *(const v8h*)(rp + k0);   // ds_load_b128
  v8h hi = *(const v8h*)(rp + k1);   // ds_load_b128
  return cat8(lo, hi);
}

// B-fragment (32x16 f16, KxN) from LDS matrix stored row-major [N][64(K)] halves.
// lanes 0-15: N=n0+lane, elems 0..7 = K kc*32+0..7, elems 8..15 = K kc*32+16..23 ; lanes 16-31: N=n0+lane-16, elems 0..7 = K kc*32+8..15, elems 8..15 = K kc*32+24..31
static __device__ __forceinline__ v16h loadB_lds(const _Float16* w, int n0, int kc, int lh, int hs) {
  const _Float16* rp = w + (n0 + lh) * HDIM + kc * 32 + 8 * hs;
  v8h lo = *(const v8h*)(rp);        // ds_load_b128
  v8h hi = *(const v8h*)(rp + 16);   // ds_load_b128
  return cat8(lo, hi);
}

// A-fragment built from a global f32 row-major [*][64] matrix (h0, W_ih)
static __device__ __forceinline__ v16h loadA_g32(const float* __restrict__ src, int row0,
                                                 int kbase, int lh, int hs) {
  const float* rp = src + (size_t)(row0 + lh) * HDIM;
  const int k0 = kbase + 8 * hs;
  const int k1 = kbase + 16 + 8 * hs;
  v16h a;
#pragma unroll
  for (int i = 0; i < 8; ++i) {
    a[i]     = (_Float16)rp[k0 + i];
    a[8 + i] = (_Float16)rp[k1 + i];
  }
  return a;
}

// Hardware tanh if the toolchain exposes it (V_TANH_F32 is a CDNA5 TRANS op);
// otherwise a cheap exp-based fallback. Guarded so compilation never breaks.
#if defined(__has_builtin)
#if __has_builtin(__builtin_amdgcn_tanhf)
#define HAVE_HW_TANH 1
#endif
#endif
static __device__ __forceinline__ float fast_tanh(float x) {
#ifdef HAVE_HW_TANH
  return __builtin_amdgcn_tanhf(x);
#else
  float e = __expf(-2.0f * x);
  return (1.0f - e) * __builtin_amdgcn_rcpf(1.0f + e);
#endif
}
static __device__ __forceinline__ float sigm(float x) {
  return __builtin_fmaf(0.5f, fast_tanh(0.5f * x), 0.5f);
}

// One 16x16 gate tile: bias + x1@Wfold.T + h0@Whh.T   (4 WMMAs)
static __device__ __forceinline__ v8f gate_tile(const _Float16* sWf, const _Float16* sWhh,
                                                const float* sBg, int n,
                                                v16h ax0, v16h ax1, v16h ah0, v16h ah1,
                                                int lh, int hs) {
  v8f c = bcast8(sBg[n * 16 + lh]);
  c = wmma_f16(ax0, loadB_lds(sWf,  n * 16, 0, lh, hs), c);
  c = wmma_f16(ax1, loadB_lds(sWf,  n * 16, 1, lh, hs), c);
  c = wmma_f16(ah0, loadB_lds(sWhh, n * 16, 0, lh, hs), c);
  c = wmma_f16(ah1, loadB_lds(sWhh, n * 16, 1, lh, hs), c);
  return c;
}

__global__ __launch_bounds__(256)
void l2l_opt_kernel(const float* __restrict__ g,   const float* __restrict__ p,
                    const float* __restrict__ mom, const float* __restrict__ h0,
                    const float* __restrict__ c0,
                    const float* __restrict__ W1,  const float* __restrict__ b1,
                    const float* __restrict__ W2,  const float* __restrict__ b2,
                    const float* __restrict__ Wih, const float* __restrict__ Whh,
                    const float* __restrict__ bih, const float* __restrict__ bhh,
                    const float* __restrict__ Wout, const float* __restrict__ bout,
                    float* __restrict__ out, int ntiles)
{
  __shared__ _Float16 sW1[HDIM * 8];        // 64 x 8 (K padded 3->8, rest 0)
  __shared__ _Float16 sZero[16];            // zero stub for lanes 16..31 B-frag reads
  __shared__ _Float16 sW2t[HDIM * HDIM];    // W2 transposed: [n(in)][k(out)]  (preload only)
  __shared__ _Float16 sWfold[NGATE * HDIM]; // Wih @ W2 : [gate(256)][hidden(64)]
  __shared__ _Float16 sWhh[NGATE * HDIM];   // 256 x 64
  __shared__ float    sB1[HDIM], sBg[NGATE], sWo[HDIM];
  __shared__ _Float16 sX[8][16 * HDIM];     // per-wave activation staging tile

  const int lane = threadIdx.x & 31;
  const int wave = threadIdx.x >> 5;
  const int lh   = lane & 15;   // index within 16-lane half
  const int hs   = lane >> 4;   // which half of the wave

  // ---- phase 1: raw weight preload + f32->f16 convert ----
  for (int i = threadIdx.x; i < NGATE * HDIM; i += blockDim.x)
    sWhh[i] = (_Float16)Whh[i];
  for (int i = threadIdx.x; i < HDIM * HDIM; i += blockDim.x) {
    const int n = i >> 6, k = i & 63;
    sW2t[i] = (_Float16)W2[k * HDIM + n];   // transpose: [n][k]
  }
  for (int i = threadIdx.x; i < HDIM; i += blockDim.x) {
#pragma unroll
    for (int k = 0; k < 8; ++k)
      sW1[i * 8 + k] = (k < 3) ? (_Float16)W1[i * 3 + k] : (_Float16)0.0f;
    sB1[i] = b1[i];
    sWo[i] = Wout[i];
  }
  if (threadIdx.x < 16) sZero[threadIdx.x] = (_Float16)0.0f;
  // gate bias with folded b2 contribution: bih + bhh + b2 @ Wih^T
  for (int i = threadIdx.x; i < NGATE; i += blockDim.x) {
    float acc = bih[i] + bhh[i];
#pragma unroll 8
    for (int k = 0; k < HDIM; ++k) acc = __builtin_fmaf(b2[k], Wih[i * HDIM + k], acc);
    sBg[i] = acc;
  }
  __syncthreads();

  // ---- phase 2: fold Wfold = Wih @ W2 with WMMA (no ReLU between L2 and gates) ----
  // Each wave computes 2 of the 16 M-tiles (M = gate rows).
#pragma unroll
  for (int t = 0; t < 2; ++t) {
    const int mt = wave * 2 + t;
    v16h aw0 = loadA_g32(Wih, mt * 16, 0, lh, hs);
    v16h aw1 = loadA_g32(Wih, mt * 16, 32, lh, hs);
#pragma unroll
    for (int n = 0; n < 4; ++n) {
      v8f c = {};
      c = wmma_f16(aw0, loadB_lds(sW2t, n * 16, 0, lh, hs), c);
      c = wmma_f16(aw1, loadB_lds(sW2t, n * 16, 1, lh, hs), c);
#pragma unroll
      for (int r = 0; r < 8; ++r)
        sWfold[(mt * 16 + r + 8 * hs) * HDIM + n * 16 + lh] = (_Float16)c[r];
    }
  }
  __syncthreads();

  // ---- phase 3: stream the batch, 16 rows per wave-tile ----
  const float bo = bout[0];
  _Float16* sx = &sX[wave][0];
  const v8h hz = {};

  const int waves_total = (int)gridDim.x * (blockDim.x >> 5);
  for (int tile = (int)blockIdx.x * (blockDim.x >> 5) + wave; tile < ntiles;
       tile += waves_total) {
    const int base = tile * 16;

    // ===== layer 1: feats(3) -> 64, zero-padded K=32 WMMA (branch-free) =====
    const float gv = g[base + lh];
    const float pv = p[base + lh];
    const float mv = __builtin_fmaf(0.9f, mom[base + lh], gv);  // momentum update
    v16h a1 = {};
    a1[0] = (_Float16)(hs ? 0.0f : gv);
    a1[1] = (_Float16)(hs ? 0.0f : pv);
    a1[2] = (_Float16)(hs ? 0.0f : mv);
#pragma unroll
    for (int n = 0; n < 4; ++n) {
      // lanes 0..15 read the padded W1 row; lanes 16..31 read the zero stub
      const _Float16* rp = hs ? sZero : &sW1[(n * 16 + lh) * 8];
      v16h bf = cat8(*(const v8h*)rp, hz);
      v8f c = bcast8(sB1[n * 16 + lh]);
      c = wmma_f16(a1, bf, c);
      // ReLU + stage to LDS (C layout -> row-major f16 tile)
#pragma unroll
      for (int r = 0; r < 8; ++r)
        sx[(r + 8 * hs) * HDIM + n * 16 + lh] = (_Float16)fmaxf(c[r], 0.0f);
    }
    asm volatile("s_wait_dscnt 0x0" ::: "memory");  // per-wave staging visible
    v16h ax0 = loadA_lds(sx, 0, lh, hs);
    v16h ax1 = loadA_lds(sx, 32, lh, hs);

    // h0 A-fragments straight from HBM (row-major matches A layout)
    v16h ah0 = loadA_g32(h0, base, 0, lh, hs);
    v16h ah1 = loadA_g32(h0, base, 32, lh, hs);

    // ===== LSTM gates: x1@Wfold^T + h0@Whh^T, fused nonlinearity =====
    v8f accO = {};
#pragma unroll
    for (int j = 0; j < 4; ++j) {   // 16-col groups of the hidden dim
      v8f cI = gate_tile(sWfold, sWhh, sBg, j,      ax0, ax1, ah0, ah1, lh, hs);
      v8f cF = gate_tile(sWfold, sWhh, sBg, j + 4,  ax0, ax1, ah0, ah1, lh, hs);
      v8f cG = gate_tile(sWfold, sWhh, sBg, j + 8,  ax0, ax1, ah0, ah1, lh, hs);
      v8f cO = gate_tile(sWfold, sWhh, sBg, j + 12, ax0, ax1, ah0, ah1, lh, hs);
      const float wv = sWo[j * 16 + lh];
#pragma unroll
      for (int r = 0; r < 8; ++r) {
        const float c0v = c0[(size_t)(base + r + 8 * hs) * HDIM + j * 16 + lh];
        const float iv = sigm(cI[r]);
        const float fv = sigm(cF[r]);
        const float gt = fast_tanh(cG[r]);
        const float ov = sigm(cO[r]);
        const float c1 = __builtin_fmaf(fv, c0v, iv * gt);
        const float h1 = ov * fast_tanh(c1);
        accO[r] = __builtin_fmaf(h1, wv, accO[r]);  // partial of h1 @ W_out^T
      }
    }

    // ===== output: reduce over the 16-lane N groups =====
#pragma unroll
    for (int off = 1; off < 16; off <<= 1)
#pragma unroll
      for (int r = 0; r < 8; ++r)
        accO[r] += __shfl_xor(accO[r], off, 32);
    if (lh == 0) {   // lane 0 -> rows 0..7, lane 16 -> rows 8..15
#pragma unroll
      for (int r = 0; r < 8; ++r)
        out[base + 8 * hs + r] = accO[r] + bo;
    }
  }
}

extern "C" void kernel_launch(void* const* d_in, const int* in_sizes, int n_in,
                              void* d_out, int out_size, void* d_ws, size_t ws_size,
                              hipStream_t stream) {
  const float* g    = (const float*)d_in[0];
  const float* p    = (const float*)d_in[1];
  const float* mom  = (const float*)d_in[2];
  const float* h0   = (const float*)d_in[3];
  const float* c0   = (const float*)d_in[4];
  const float* W1   = (const float*)d_in[5];
  const float* b1   = (const float*)d_in[6];
  const float* W2   = (const float*)d_in[7];
  const float* b2   = (const float*)d_in[8];
  const float* Wih  = (const float*)d_in[9];
  const float* Whh  = (const float*)d_in[10];
  const float* bih  = (const float*)d_in[11];
  const float* bhh  = (const float*)d_in[12];
  const float* Wout = (const float*)d_in[13];
  const float* bout = (const float*)d_in[14];
  float* out = (float*)d_out;

  const int B = in_sizes[0];        // D == 1, gradients is [B]
  const int ntiles = B / 16;        // 16 batch rows per wave-tile
  int blocks = (ntiles + 7) / 8;    // 8 waves (256 threads) per block
  if (blocks > 1024) blocks = 1024; // grid-stride; amortize weight preload/fold
  l2l_opt_kernel<<<blocks, 256, 0, stream>>>(g, p, mom, h0, c0,
                                             W1, b1, W2, b2,
                                             Wih, Whh, bih, bhh,
                                             Wout, bout, out, ntiles);
}
